// GNN_28243704939250
// MI455X (gfx1250) — matrix operand order repaired and checked by reading, not compiled, not run on hardware
//
#include <hip/hip_runtime.h>
#include <hip/hip_bf16.h>
#include <math.h>

// ---------------------------------------------------------------------------
// MI455X (gfx1250) implementation.
// Core compute: v_wmma_f32_16x16x32_f16 (f16 in, f32 accumulate), 128x64 block
// tiles, 8 WMMAs per wave per K-step, double-buffered LDS staged with
// GLOBAL_LOAD_ASYNC_TO_LDS_B128 (ASYNCcnt) when the toolchain exposes it.
// The ~200MB working set is L2-resident (192MB L2); diffusion GEMMs are
// compute-bound -> WMMA. Sinkhorn is bandwidth-bound streaming.
// ---------------------------------------------------------------------------

typedef __attribute__((ext_vector_type(16))) _Float16 v16h;
typedef __attribute__((ext_vector_type(8)))  _Float16 v8h;
typedef __attribute__((ext_vector_type(8)))  float    v8f;
typedef __attribute__((ext_vector_type(4)))  int      v4i;

#define B_        16
#define N_        1024
#define IN_DIM_   128
#define HID_      256
#define OUT_DIM_  1024
#define TAU_      0.1f
#define NOISE_    0.05f
#define TANH_SC_  40.0f
#define N_ITER_   20

#define TM_ 128
#define TN_ 64

#if defined(__gfx1250__) && __has_builtin(__builtin_amdgcn_global_load_async_to_lds_b128) && __has_builtin(__builtin_amdgcn_s_wait_asynccnt)
#define HAVE_ASYNC_LDS 1
#else
#define HAVE_ASYNC_LDS 0
#endif

typedef __attribute__((address_space(1))) v4i* gv4i_p;   // global int4*
typedef __attribute__((address_space(3))) v4i* lv4i_p;   // LDS int4*

// 16B global -> LDS copy. Async path on gfx1250 (no VGPR round-trip, ASYNCcnt),
// falls back to a vector load/store pair. Generic LDS addr low 32 bits == LDS
// offset (flat aperture rule); global addr is identity in flat space.
__device__ __forceinline__ void copy16_g2l(const void* g, void* l) {
#if HAVE_ASYNC_LDS
  __builtin_amdgcn_global_load_async_to_lds_b128(
      (gv4i_p)(unsigned long long)g,
      (lv4i_p)(unsigned)(unsigned long long)l, 0, 0);
#else
  *(v8h*)l = *(const v8h*)g;
#endif
}
__device__ __forceinline__ void async_fence() {
#if HAVE_ASYNC_LDS
  __builtin_amdgcn_s_wait_asynccnt(0);
#endif
}

// ------------------------------- utility -----------------------------------

__global__ void k_f32_to_f16(const float* __restrict__ in, _Float16* __restrict__ out, int n) {
  int i = blockIdx.x * 256 + threadIdx.x;
  if (i < n) out[i] = (_Float16)in[i];
}

// per-row degree sums of adj: dis = (rowsum+1)^-1/2 (A+I), rs = rowsum (for P)
__global__ void k_degrees(const float* __restrict__ adj, float* __restrict__ dis,
                          float* __restrict__ rs) {
  __shared__ float red[256];
  const int row = blockIdx.x;                 // b*N + i
  const float* p = adj + (size_t)row * N_;
  float s = 0.f;
  for (int j = threadIdx.x; j < N_; j += 256) s += p[j];
  red[threadIdx.x] = s; __syncthreads();
  for (int o = 128; o > 0; o >>= 1) {
    if (threadIdx.x < o) red[threadIdx.x] += red[threadIdx.x + o];
    __syncthreads();
  }
  if (threadIdx.x == 0) {
    float t = red[0];
    rs[row]  = t;
    dis[row] = rsqrtf(t + 1.0f);
  }
}

// A_hat = D^-1/2 (A+I) D^-1/2 ;  PL = 0.5*(I + D^-1 A)  (lazy walk, 1 GEMM/apply)
__global__ void k_build(const float* __restrict__ adj, const float* __restrict__ dis,
                        const float* __restrict__ rs, _Float16* __restrict__ Ahat,
                        _Float16* __restrict__ PL) {
  size_t idx = (size_t)blockIdx.x * 256 + threadIdx.x;
  if (idx >= (size_t)B_ * N_ * N_) return;
  int b   = (int)(idx >> 20);
  int rem = (int)(idx & ((1u << 20) - 1));
  int i = rem >> 10, j = rem & 1023;
  float a = adj[idx];
  float dg = (i == j) ? 1.f : 0.f;
  float ah = dis[b * N_ + i] * dis[b * N_ + j] * (a + dg);
  float pl = 0.5f * a / rs[b * N_ + i] + 0.5f * dg;
  Ahat[idx] = (_Float16)ah;
  PL[idx]   = (_Float16)pl;
}

// ------------------------------ WMMA GEMM ----------------------------------
// C[M,Ncol] = epi( A[M,K] x op(B) + bias ), f16 inputs, f32 accumulate.
// transB=1: B is [Ncol,K] row-major (PyTorch weight); transB=0: B is [K,Ncol].
// Block tile 128x64, 128 threads (4 waves); wave computes 32x64 = 8 WMMA tiles.
// Double-buffered LDS, async global->LDS staging. epi: 0=none,1=leaky,2=tanh*40
__launch_bounds__(128)
__global__ void k_gemm_wmma(const _Float16* __restrict__ A, long long sAb, int lda,
                            const _Float16* __restrict__ Bm, long long sBb, int ldb,
                            int transB, const float* __restrict__ bias,
                            float* __restrict__ C, long long sCb, int ldc,
                            _Float16* __restrict__ C16, long long sC16b, int ldc16,
                            int M, int Ncol, int K, int epi) {
  __shared__ __align__(16) _Float16 sA [2][TM_][40];  // 80B row stride: 16B aligned
  __shared__ __align__(16) _Float16 sBT[2][TN_][40];  // [col][k]
  const int tid  = threadIdx.x;
  const int lane = tid & 31;
  const int wave = tid >> 5;
  const int m0 = blockIdx.y * TM_;
  const int n0 = blockIdx.x * TN_;
  const int z  = blockIdx.z;
  A  += (size_t)z * sAb;
  Bm += (size_t)z * sBb;

  auto stage = [&](int buf, int k0) {
    // A tile 128x32: 512 contiguous 16B chunks
    #pragma unroll
    for (int c = 0; c < 4; ++c) {
      int idx = tid + c * 128;
      int row = idx >> 2, kc = (idx & 3) * 8;
      copy16_g2l(&A[(size_t)(m0 + row) * lda + k0 + kc], &sA[buf][row][kc]);
    }
    if (transB) {
      #pragma unroll
      for (int c = 0; c < 2; ++c) {
        int idx = tid + c * 128;
        int col = idx >> 2, kc = (idx & 3) * 8;
        copy16_g2l(&Bm[(size_t)(n0 + col) * ldb + k0 + kc], &sBT[buf][col][kc]);
      }
    } else {
      for (int e = tid; e < TN_ * 32; e += 128) {
        int k = e >> 6, col = e & 63;               // coalesced over col
        sBT[buf][col][k] = Bm[(size_t)(k0 + k) * ldb + n0 + col];
      }
    }
  };

  v8f acc[2][4] = {};
  stage(0, 0);
  async_fence();
  __syncthreads();

  int buf = 0;
  for (int k0 = 0; k0 < K; k0 += 32) {
    if (k0 + 32 < K) stage(buf ^ 1, k0 + 32);       // prefetch next tile

    // A fragments: lanes<16 K=0..15 ; lanes>=16 K=16..31 (matches B)
    const int r0  = wave * 32 + (lane & 15);
    const int kbA = (lane & 16) ? 16 : 0;
    v16h af0, af1;
    #pragma unroll
    for (int i = 0; i < 8; ++i) {
      af0[i]     = sA[buf][r0][kbA + i];
      af0[8 + i] = sA[buf][r0][kbA + 8 + i];
      af1[i]     = sA[buf][r0 + 16][kbA + i];
      af1[8 + i] = sA[buf][r0 + 16][kbA + 8 + i];
    }
    // B fragments: lanes<16 K=0..15 ; lanes>=16 K=16..31 (col = lane&15)
    const int kbB = (lane & 16) ? 16 : 0;
    #pragma unroll
    for (int t = 0; t < 4; ++t) {
      const int col = t * 16 + (lane & 15);
      v16h bf;
      #pragma unroll
      for (int i = 0; i < 16; ++i) bf[i] = sBT[buf][col][kbB + i];
      acc[0][t] = __builtin_amdgcn_wmma_f32_16x16x32_f16(
          false, af0, false, bf, (short)0, acc[0][t], false, false);
      acc[1][t] = __builtin_amdgcn_wmma_f32_16x16x32_f16(
          false, af1, false, bf, (short)0, acc[1][t], false, false);
    }
    async_fence();     // own async copies complete before making buffer visible
    __syncthreads();   // everyone done reading buf & staging buf^1
    buf ^= 1;
  }

  // epilogue: C/D layout VGPR j -> M=j (lanes<16) or M=8+j (lanes>=16)
  const int colInLane = lane & 15;
  C += (size_t)z * sCb;
  if (C16) C16 += (size_t)z * sC16b;
  #pragma unroll
  for (int h = 0; h < 2; ++h) {
    const int rowBase = m0 + wave * 32 + h * 16 + ((lane & 16) ? 8 : 0);
    #pragma unroll
    for (int t = 0; t < 4; ++t) {
      int col = n0 + t * 16 + colInLane;
      float bv = bias ? bias[col] : 0.f;
      #pragma unroll
      for (int j = 0; j < 8; ++j) {
        float v = acc[h][t][j] + bv;
        if (epi == 1)      v = (v > 0.f) ? v : 0.01f * v;
        else if (epi == 2) v = tanhf(v) * TANH_SC_;
        int rowg = rowBase + j;
        C[(size_t)rowg * ldc + col] = v;
        if (C16) C16[(size_t)rowg * ldc16 + col] = (_Float16)v;
      }
    }
  }
}

// -------------------- attention + scattering differences -------------------
// channels: c0=leaky(gcn) (precomputed in GEMM epi), c1=|X-p1|, c2=|p1-p2|,
// c3=|p2-p4|. e_c = sum relu(X)*a[0:H] + sum relu(ch_c)*a[H:2H]; softmax over
// 4 channels; h_prime = mean_c att_c*ch_c -> f16 for the next GEMM.
__global__ void k_attn(const float* __restrict__ X, const float* __restrict__ g1,
                       const float* __restrict__ p1, const float* __restrict__ p2,
                       const float* __restrict__ p4, const float* __restrict__ avec,
                       _Float16* __restrict__ hp16) {
  __shared__ float red[256];
  __shared__ float att[4];
  const int row = blockIdx.x;
  const int k   = threadIdx.x;      // blockDim.x == HID_
  const size_t off = (size_t)row * HID_ + k;
  const float xk  = X[off];
  const float c0  = g1[off];
  const float q1 = p1[off], q2 = p2[off], q4 = p4[off];
  const float c1 = fabsf(xk - q1);
  const float c2 = fabsf(q1 - q2);
  const float c3 = fabsf(q2 - q4);
  const float ax = avec[k], ah = avec[HID_ + k];
  float part[5];
  part[0] = fmaxf(c0, 0.f) * ah;
  part[1] = c1 * ah;                 // c1..c3 >= 0: relu is identity
  part[2] = c2 * ah;
  part[3] = c3 * ah;
  part[4] = fmaxf(xk, 0.f) * ax;
  float sums[5];
  for (int q = 0; q < 5; ++q) {
    red[k] = part[q]; __syncthreads();
    for (int o = 128; o > 0; o >>= 1) {
      if (k < o) red[k] += red[k + o];
      __syncthreads();
    }
    sums[q] = red[0]; __syncthreads();
  }
  if (k == 0) {
    float ex = sums[4];
    float e0 = sums[0] + ex, e1 = sums[1] + ex, e2 = sums[2] + ex, e3 = sums[3] + ex;
    float m = fmaxf(fmaxf(e0, e1), fmaxf(e2, e3));
    float w0 = __expf(e0 - m), w1 = __expf(e1 - m), w2 = __expf(e2 - m), w3 = __expf(e3 - m);
    float s = w0 + w1 + w2 + w3;
    att[0] = w0 / s; att[1] = w1 / s; att[2] = w2 / s; att[3] = w3 / s;
  }
  __syncthreads();
  float hp = 0.25f * (att[0] * c0 + att[1] * c1 + att[2] * c2 + att[3] * c3);
  hp16[off] = (_Float16)hp;
}

// ------------------------------- sinkhorn ----------------------------------

__device__ __forceinline__ float hash_u01(unsigned x) {
  x ^= x >> 17; x *= 0xed5ad4bbu;
  x ^= x >> 11; x *= 0xac4c1b51u;
  x ^= x >> 15; x *= 0x31848babu;
  x ^= x >> 14;
  return ((float)(x >> 8) + 0.5f) * (1.0f / 16777216.0f);
}

__global__ void k_z0(const float* __restrict__ logits, float* __restrict__ Z,
                     float* __restrict__ z0out, int n) {
  int i = blockIdx.x * 256 + threadIdx.x;
  if (i >= n) return;
  float u = hash_u01((unsigned)i * 2654435761u + 0x9E3779B9u);
  float noise = -logf(-logf(u + 1e-20f) + 1e-20f) * NOISE_;
  float z = (logits[i] + noise) / TAU_;
  Z[i] = z;
  z0out[i] = z;
}

__global__ void k_lse_rows(float* __restrict__ Z) {
  __shared__ float red[256];
  const int row = blockIdx.x;
  float* p = Z + (size_t)row * OUT_DIM_;
  const int t = threadIdx.x;
  float v[4]; float m = -1e30f;
  #pragma unroll
  for (int q = 0; q < 4; ++q) { v[q] = p[t + 256 * q]; m = fmaxf(m, v[q]); }
  red[t] = m; __syncthreads();
  for (int o = 128; o > 0; o >>= 1) { if (t < o) red[t] = fmaxf(red[t], red[t + o]); __syncthreads(); }
  float M = red[0]; __syncthreads();
  float s = 0.f;
  #pragma unroll
  for (int q = 0; q < 4; ++q) s += __expf(v[q] - M);
  red[t] = s; __syncthreads();
  for (int o = 128; o > 0; o >>= 1) { if (t < o) red[t] += red[t + o]; __syncthreads(); }
  float lse = M + logf(red[0]);
  #pragma unroll
  for (int q = 0; q < 4; ++q) p[t + 256 * q] = v[q] - lse;
}

// online LSE down columns, one column per thread, coalesced row streaming
__global__ void k_lse_cols(float* __restrict__ Z) {
  const int col = blockIdx.x * 256 + threadIdx.x;
  float* base = Z + (size_t)blockIdx.y * N_ * OUT_DIM_;
  float m = -1e30f, s = 0.f;
  for (int i = 0; i < N_; ++i) {
    float x = base[(size_t)i * OUT_DIM_ + col];
    if (x > m) { s = s * __expf(m - x) + 1.f; m = x; }
    else       { s += __expf(x - m); }
  }
  float lse = m + logf(s);
  for (int i = 0; i < N_; ++i) base[(size_t)i * OUT_DIM_ + col] -= lse;
}

__global__ void k_exp(const float* __restrict__ Z, float* __restrict__ P, int n) {
  int i = blockIdx.x * 256 + threadIdx.x;
  if (i < n) P[i] = __expf(Z[i]);
}

__global__ void k_rownorm(float* __restrict__ P) {
  __shared__ float red[256];
  const int row = blockIdx.x;
  float* p = P + (size_t)row * OUT_DIM_;
  const int t = threadIdx.x;
  float v[4]; float s = 0.f;
  #pragma unroll
  for (int q = 0; q < 4; ++q) { v[q] = p[t + 256 * q]; s += v[q]; }
  red[t] = s; __syncthreads();
  for (int o = 128; o > 0; o >>= 1) { if (t < o) red[t] += red[t + o]; __syncthreads(); }
  float inv = 1.f / red[0];
  #pragma unroll
  for (int q = 0; q < 4; ++q) p[t + 256 * q] = v[q] * inv;
}

__global__ void k_colnorm(float* __restrict__ P) {
  const int col = blockIdx.x * 256 + threadIdx.x;
  float* base = P + (size_t)blockIdx.y * N_ * OUT_DIM_;
  float s = 0.f;
  for (int i = 0; i < N_; ++i) s += base[(size_t)i * OUT_DIM_ + col];
  float inv = 1.f / s;
  for (int i = 0; i < N_; ++i) base[(size_t)i * OUT_DIM_ + col] *= inv;
}

// ------------------------------- driver ------------------------------------

extern "C" void kernel_launch(void* const* d_in, const int* in_sizes, int n_in,
                              void* d_out, int out_size, void* d_ws, size_t ws_size,
                              hipStream_t stream) {
  (void)in_sizes; (void)n_in; (void)out_size; (void)ws_size;
  const float* X       = (const float*)d_in[0];
  const float* adj     = (const float*)d_in[1];
  const float* w_in    = (const float*)d_in[2];
  const float* b_in    = (const float*)d_in[3];
  const float* conv_w1 = (const float*)d_in[4];
  const float* conv_b1 = (const float*)d_in[5];
  const float* conv_w2 = (const float*)d_in[6];
  const float* conv_b2 = (const float*)d_in[7];
  const float* conv_a  = (const float*)d_in[8];
  const float* mlp1_w  = (const float*)d_in[9];
  const float* mlp1_b  = (const float*)d_in[10];
  const float* mlp2_w  = (const float*)d_in[11];
  const float* mlp2_b  = (const float*)d_in[12];

  const size_t NN = (size_t)B_ * N_ * N_;   // 16M
  const size_t BN = (size_t)B_ * N_;        // 16384
  const size_t FH = BN * HID_;              // 4M

  char* wsb = (char*)d_ws;
  size_t cur = 0;
  auto alloc = [&](size_t bytes) -> void* {
    cur = (cur + 255) & ~(size_t)255;
    void* p = wsb + cur;
    cur += bytes;
    return p;
  };

  _Float16* Ahat16 = (_Float16*)alloc(NN * 2);
  _Float16* PL16   = (_Float16*)alloc(NN * 2);
  _Float16* X16    = (_Float16*)alloc(BN * IN_DIM_ * 2);
  _Float16* win16  = (_Float16*)alloc((size_t)HID_ * IN_DIM_ * 2);
  _Float16* cw116  = (_Float16*)alloc((size_t)2 * HID_ * HID_ * 2);
  _Float16* cw216  = (_Float16*)alloc((size_t)2 * HID_ * HID_ * 2);
  _Float16* m1w16  = (_Float16*)alloc((size_t)HID_ * 3 * HID_ * 2);
  _Float16* m2w16  = (_Float16*)alloc((size_t)OUT_DIM_ * HID_ * 2);
  float*    dis    = (float*)alloc(BN * 4);
  float*    rs     = (float*)alloc(BN * 4);
  _Float16* hcat16 = (_Float16*)alloc(BN * 3 * HID_ * 2);   // [BN,768] f16 concat
  float* h0_32 = (float*)alloc(FH * 4);
  float* h1_32 = (float*)alloc(FH * 4);
  float* h2_32 = (float*)alloc(FH * 4);
  float* g1_32 = (float*)alloc(FH * 4);
  float* p32[4]; _Float16* p16[4];
  for (int i = 0; i < 4; ++i) {
    p32[i] = (float*)alloc(FH * 4);
    p16[i] = (_Float16*)alloc(FH * 2);
  }
  _Float16* hp16 = (_Float16*)alloc(FH * 2);
  float*    t32  = (float*)alloc(FH * 4);
  _Float16* t16  = (_Float16*)alloc(FH * 2);
  float*    mh32 = (float*)alloc(FH * 4);
  _Float16* mh16 = (_Float16*)alloc(FH * 2);
  float*  logits = (float*)alloc(BN * OUT_DIM_ * 4);  // reused as Z in-place

  auto cvt = [&](const float* in, _Float16* out, size_t n) {
    k_f32_to_f16<<<dim3((unsigned)((n + 255) / 256)), dim3(256), 0, stream>>>(in, out, (int)n);
  };
  cvt(X, X16, BN * IN_DIM_);
  cvt(w_in, win16, (size_t)HID_ * IN_DIM_);
  cvt(conv_w1, cw116, (size_t)2 * HID_ * HID_);
  cvt(conv_w2, cw216, (size_t)2 * HID_ * HID_);
  cvt(mlp1_w, m1w16, (size_t)HID_ * 3 * HID_);
  cvt(mlp2_w, m2w16, (size_t)OUT_DIM_ * HID_);

  k_degrees<<<dim3((unsigned)BN), dim3(256), 0, stream>>>(adj, dis, rs);
  k_build<<<dim3((unsigned)((NN + 255) / 256)), dim3(256), 0, stream>>>(adj, dis, rs, Ahat16, PL16);

  auto gemm = [&](const _Float16* A, long long sAb, int lda,
                  const _Float16* Bm, long long sBb, int ldb, int transB,
                  const float* bias, float* C, long long sCb, int ldc,
                  _Float16* C16, long long sC16b, int ldc16,
                  int M, int Ncol, int K, int epi, int batches) {
    dim3 g(Ncol / TN_, M / TM_, batches), blk(128);
    k_gemm_wmma<<<g, blk, 0, stream>>>(A, sAb, lda, Bm, sBb, ldb, transB, bias,
                                       C, sCb, ldc, C16, sC16b, ldc16, M, Ncol, K, epi);
  };

  // h0 = X @ w_in^T + b_in  (no activation); f16 copy into concat cols [0,256)
  gemm(X16, 0, IN_DIM_, win16, 0, IN_DIM_, 1, b_in,
       h0_32, 0, HID_, hcat16, 0, 3 * HID_, (int)BN, HID_, IN_DIM_, 0, 1);

  const long long sFeat16 = (long long)N_ * 3 * HID_;   // batch stride inside hcat
  const long long sFeat32 = (long long)N_ * HID_;
  for (int l = 0; l < 2; ++l) {
    const float*    hin32 = (l == 0) ? h0_32 : h1_32;
    const _Float16* hin16 = hcat16 + (size_t)l * HID_;  // ld = 768
    // gcn channel: leaky(A_hat @ h), batched over B_
    gemm(Ahat16, (long long)N_ * N_, N_, hin16, sFeat16, 3 * HID_, 0, nullptr,
         g1_32, sFeat32, HID_, nullptr, 0, 0, N_, HID_, N_, 1, B_);
    // lazy-walk powers p1..p4 (each one GEMM with PL = 0.5(I+P))
    gemm(PL16, (long long)N_ * N_, N_, hin16, sFeat16, 3 * HID_, 0, nullptr,
         p32[0], sFeat32, HID_, p16[0], sFeat32, HID_, N_, HID_, N_, 0, B_);
    for (int s = 1; s < 4; ++s)
      gemm(PL16, (long long)N_ * N_, N_, p16[s - 1], sFeat32, HID_, 0, nullptr,
           p32[s], sFeat32, HID_, p16[s], sFeat32, HID_, N_, HID_, N_, 0, B_);
    // attention over {gcn, |X-p1|, |p1-p2|, |p2-p4|} -> h_prime (f16)
    k_attn<<<dim3((unsigned)BN), dim3(HID_), 0, stream>>>(
        hin32, g1_32, p32[0], p32[1], p32[3], conv_a + (size_t)l * 2 * HID_, hp16);
    // t = leaky(h_prime @ w1^T + b1)
    gemm(hp16, 0, HID_, cw116 + (size_t)l * HID_ * HID_, 0, HID_, 1,
         conv_b1 + (size_t)l * HID_, t32, 0, HID_, t16, 0, HID_,
         (int)BN, HID_, HID_, 1, 1);
    // h_out = leaky(t @ w2^T + b2); f16 into concat cols [(l+1)*256, ...)
    float* hout32 = (l == 0) ? h1_32 : h2_32;
    gemm(t16, 0, HID_, cw216 + (size_t)l * HID_ * HID_, 0, HID_, 1,
         conv_b2 + (size_t)l * HID_, hout32, 0, HID_,
         hcat16 + (size_t)(l + 1) * HID_, 0, 3 * HID_,
         (int)BN, HID_, HID_, 1, 1);
  }

  // mlp1: leaky(hcat @ mlp1_w^T + b)
  gemm(hcat16, 0, 3 * HID_, m1w16, 0, 3 * HID_, 1, mlp1_b,
       mh32, 0, HID_, mh16, 0, HID_, (int)BN, HID_, 3 * HID_, 1, 1);
  // mlp2: tanh(mh @ mlp2_w^T + b) * 40 -> logits
  gemm(mh16, 0, HID_, m2w16, 0, HID_, 1, mlp2_b,
       logits, 0, OUT_DIM_, nullptr, 0, 0, (int)BN, OUT_DIM_, HID_, 2, 1);

  // gumbel-sinkhorn
  float* out = (float*)d_out;
  float* outP  = out;               // first 16M floats
  float* outZ0 = out + NN;          // second 16M floats
  const int nEl = (int)NN;
  k_z0<<<dim3((unsigned)(NN / 256)), dim3(256), 0, stream>>>(logits, logits, outZ0, nEl);
  for (int it = 0; it < N_ITER_; ++it) {
    k_lse_rows<<<dim3((unsigned)BN), dim3(256), 0, stream>>>(logits);
    k_lse_cols<<<dim3(OUT_DIM_ / 256, B_), dim3(256), 0, stream>>>(logits);
  }
  k_exp<<<dim3((unsigned)(NN / 256)), dim3(256), 0, stream>>>(logits, outP, nEl);
  k_rownorm<<<dim3((unsigned)BN), dim3(256), 0, stream>>>(outP);
  k_colnorm<<<dim3(OUT_DIM_ / 256, B_), dim3(256), 0, stream>>>(outP);
}
